// ItemModeling_45440753992065
// MI455X (gfx1250) — matrix operand order repaired and checked by reading, not compiled, not run on hardware
//
#include <hip/hip_runtime.h>

// ---------------------------------------------------------------------------
// GraphRec item-modeling, fused CDNA5 WMMA kernel.
// B=4096 blocks x 128 threads (4 wave32). Each wave owns a 16-row M-tile of
// the padded 64-row neighbor set. All row-MLP GEMMs run on
// v_wmma_f32_16x16x32_f16 (f16 in, f32 accumulate).
// ---------------------------------------------------------------------------

#define LL 50
#define DD 64

typedef __attribute__((ext_vector_type(16))) _Float16 v16h;
typedef __attribute__((ext_vector_type(8)))  float    v8f;
typedef __attribute__((ext_vector_type(4)))  float    v4f;
typedef __attribute__((ext_vector_type(4)))  _Float16 v4h;

// d_ws layout (in halfs): weights transposed to [N][K] row-major f16
#define GV1T  0       // [64][128]
#define GV2T  8192    // [64][64]
#define GV3T  12288   // [64][64]
#define ATT1T 16384   // [64][128]
#define ATT2T 24576   // [64][64]
#define WS_HALFS 28672

// ---------------------------------------------------------------------------
// Prep: fp32 weights -> f16, transposed to [N][K] so each lane's B-fragment
// is 16 contiguous halfs (two global_load_b128).
// ---------------------------------------------------------------------------
__global__ void prep_weights_k(const float* __restrict__ gv_w1,
                               const float* __restrict__ gv_w2,
                               const float* __restrict__ gv_w3,
                               const float* __restrict__ att_w1,
                               const float* __restrict__ att_w2,
                               _Float16* __restrict__ wsT) {
  int stride = gridDim.x * blockDim.x;
  int tid = blockIdx.x * blockDim.x + threadIdx.x;
  for (int i = tid; i < 64 * 128; i += stride) { int n = i >> 7, k = i & 127; wsT[GV1T  + i] = (_Float16)gv_w1 [k * 64 + n]; }
  for (int i = tid; i < 64 * 64;  i += stride) { int n = i >> 6, k = i & 63;  wsT[GV2T  + i] = (_Float16)gv_w2 [k * 64 + n]; }
  for (int i = tid; i < 64 * 64;  i += stride) { int n = i >> 6, k = i & 63;  wsT[GV3T  + i] = (_Float16)gv_w3 [k * 64 + n]; }
  for (int i = tid; i < 64 * 128; i += stride) { int n = i >> 7, k = i & 127; wsT[ATT1T + i] = (_Float16)att_w1[k * 64 + n]; }
  for (int i = tid; i < 64 * 64;  i += stride) { int n = i >> 6, k = i & 63;  wsT[ATT2T + i] = (_Float16)att_w2[k * 64 + n]; }
}

// ---------------------------------------------------------------------------
// One MLP layer for a 16-row M-tile: dst[16..][64] = act(src[16..][K] @ W + c)
//   src: LDS f16, row-major with row stride == K
//   wt : global f16, transposed [64][LDW] (only k < K used)
//   C initializer: bias[n] (global f32) or cinit[n] (LDS f32)
// A-frag: lane m = m0+(lane&15), hi lanes get K+16; chunks at +0 and +8 halfs.
// B-frag: lane n = n0+(lane&15), hi lanes get K+16; chunks at +0 and +8 halfs.
// D: lane n, VGPR r -> row m0 + r + (hi?8:0).
// ---------------------------------------------------------------------------
template <int K, int LDW>
__device__ __forceinline__ void mlp_layer(const _Float16* __restrict__ src,
                                          const _Float16* __restrict__ wt,
                                          const float* __restrict__ bias,
                                          const float* __restrict__ cinit,
                                          _Float16* __restrict__ dst,
                                          bool do_relu, int m0, int lane) {
  const int nlo = lane & 15;
  const int m   = m0 + nlo;                 // A-fragment row
  const int kA  = (lane >> 4) << 4;         // +16 halfs for hi half-wave
  const int kB  = (lane >> 4) << 4;         // +16 halfs for hi half-wave
  const int mb  = m0 + ((lane >> 4) << 3);  // D-fragment row base
#pragma unroll
  for (int nt = 0; nt < 4; ++nt) {
    const int n0 = nt * 16;
    const float cv = bias ? bias[n0 + nlo] : cinit[n0 + nlo];
    v8f c = {cv, cv, cv, cv, cv, cv, cv, cv};
#pragma unroll
    for (int k0 = 0; k0 < K; k0 += 32) {
      const _Float16* pa = src + m * K + k0 + kA;
      const _Float16* pb = wt + (n0 + nlo) * LDW + k0 + kB;
      union { v16h h; v4f f[2]; } a, b;
      a.f[0] = *(const v4f*)pa;  a.f[1] = *(const v4f*)(pa + 8);
      b.f[0] = *(const v4f*)pb;  b.f[1] = *(const v4f*)(pb + 8);
      c = __builtin_amdgcn_wmma_f32_16x16x32_f16(false, a.h, false, b.h,
                                                 (short)0, c, false, false);
    }
#pragma unroll
    for (int r = 0; r < 8; ++r) {
      float vv = c[r];
      if (do_relu) vv = vv > 0.f ? vv : 0.f;
      dst[(mb + r) * 64 + n0 + nlo] = (_Float16)vv;
    }
  }
}

// ---------------------------------------------------------------------------
// Fused kernel: one block per batch element.
// ---------------------------------------------------------------------------
__global__ __launch_bounds__(128) void item_fused_k(
    const int* __restrict__ nodes_v, const int* __restrict__ neigh_u,
    const int* __restrict__ neigh_r,
    const float* __restrict__ embed_u, const float* __restrict__ embed_i,
    const float* __restrict__ embed_r,
    const float* __restrict__ gv_b1, const float* __restrict__ gv_b2,
    const float* __restrict__ gv_b3,
    const float* __restrict__ att_w1, const float* __restrict__ att_b1,
    const float* __restrict__ att_b2,
    const float* __restrict__ att_w3, const float* __restrict__ att_b3,
    const float* __restrict__ wr1_w, const float* __restrict__ wr1_b,
    const float* __restrict__ wr2_w, const float* __restrict__ wr2_b,
    const _Float16* __restrict__ wsT,
    float* __restrict__ out) {
  __shared__ __align__(16) _Float16 X [64 * 128];  // [pt|er] concat, f16
  __shared__ __align__(16) _Float16 T1[64 * 64];
  __shared__ __align__(16) _Float16 T2[64 * 64];
  __shared__ __align__(16) _Float16 FJ[64 * 64];   // fjt (gv output), f16
  __shared__ float qjL[64], qtL[64], scL[64], muL[64], zbL[64], z2L[64], redL[2];

  const int b    = blockIdx.x;
  const int tid  = threadIdx.x;
  const int lane = tid & 31;
  const int wave = tid >> 5;
  const int m0   = wave * 16;

  // ---- Stage 1: gather pt/er rows into X (rows L..63 zero-padded) ----
  for (int i = tid; i < LL * 16; i += 128) {
    int l = i >> 4, c = (i & 15) << 2;
    int u = neigh_u[b * LL + l];
    int r = neigh_r[b * LL + l];
    v4f pu = *(const v4f*)(embed_u + (size_t)u * DD + c);
    v4f pr = *(const v4f*)(embed_r + (size_t)r * DD + c);
    *(v4h*)&X[l * 128 + c]      = __builtin_convertvector(pu, v4h);
    *(v4h*)&X[l * 128 + 64 + c] = __builtin_convertvector(pr, v4h);
  }
  for (int i = tid; i < (64 - LL) * 32; i += 128) {
    int l = LL + (i >> 5), c = (i & 31) << 2;
    v4h z = {};
    *(v4h*)&X[l * 128 + c] = z;
  }
  const int vtx = nodes_v[b];
  if (tid < 64) qjL[tid] = embed_i[(size_t)vtx * DD + tid];
  __syncthreads();

  // Row-independent half of attention layer 1: qterm = qj @ att_w1[64:] + b1
  if (tid < 64) {
    float acc = att_b1[tid];
    for (int k = 0; k < 64; ++k)
      acc = fmaf(qjL[k], att_w1[(64 + k) * 64 + tid], acc);
    qtL[tid] = acc;
  }
  __syncthreads();

  // ---- Stage 2: WMMA MLP chain (wave-local, no barriers needed) ----
  mlp_layer<128, 128>(X,  wsT + GV1T,  gv_b1,  nullptr, T1, true,  m0, lane);
  mlp_layer<64,  64 >(T1, wsT + GV2T,  gv_b2,  nullptr, T2, true,  m0, lane);
  mlp_layer<64,  64 >(T2, wsT + GV3T,  gv_b3,  nullptr, FJ, false, m0, lane);
  mlp_layer<64,  128>(FJ, wsT + ATT1T, nullptr, qtL,    T1, true,  m0, lane);
  mlp_layer<64,  64 >(T1, wsT + ATT2T, att_b2, nullptr, T2, true,  m0, lane);
  __syncthreads();

  // ---- Stage 3: attention scores (64->1 matvec per row) ----
  if (tid < 64) {
    float s = att_b3[0];
    for (int k = 0; k < 64; ++k)
      s = fmaf((float)T2[tid * 64 + k], att_w3[k], s);
    scL[tid] = s;
  }
  __syncthreads();

  // ---- Stage 4: softmax over L=50 neighbors ----
  if (tid == 0) {
    float mx = -1e30f;
    for (int l = 0; l < LL; ++l) mx = fmaxf(mx, scL[l]);
    redL[0] = mx;
  }
  __syncthreads();
  if (tid < 64) muL[tid] = (tid < LL) ? __expf(scL[tid] - redL[0]) : 0.f;
  __syncthreads();
  if (tid == 0) {
    float s = 0.f;
    for (int l = 0; l < LL; ++l) s += muL[l];
    redL[1] = 1.f / s;
  }
  __syncthreads();

  // ---- Stage 5: weighted aggregation zj = sum_l mu_l * fjt_l ----
  if (tid < 64) {
    float inv = redL[1];
    float z = 0.f;
    for (int l = 0; l < LL; ++l)
      z = fmaf(muL[l], (float)FJ[l * 64 + tid], z);
    zbL[tid] = z * inv;
  }
  __syncthreads();

  // ---- Stage 6: combine [zj, qj] @ wr1 -> relu -> @ wr2 -> relu ----
  if (tid < 64) {
    float h = wr1_b[tid];
    for (int k = 0; k < 64; ++k) h = fmaf(zbL[k], wr1_w[k * 64 + tid], h);
    for (int k = 0; k < 64; ++k) h = fmaf(qjL[k], wr1_w[(64 + k) * 64 + tid], h);
    z2L[tid] = fmaxf(h, 0.f);
  }
  __syncthreads();
  if (tid < 64) {
    float o = wr2_b[tid];
    for (int k = 0; k < 64; ++k) o = fmaf(z2L[k], wr2_w[k * 64 + tid], o);
    out[(size_t)b * DD + tid] = fmaxf(o, 0.f);
  }
}

// ---------------------------------------------------------------------------
extern "C" void kernel_launch(void* const* d_in, const int* in_sizes, int n_in,
                              void* d_out, int out_size, void* d_ws, size_t ws_size,
                              hipStream_t stream) {
  const int*   nodes_v = (const int*)d_in[0];
  const int*   neigh_u = (const int*)d_in[1];
  const int*   neigh_r = (const int*)d_in[2];
  const float* embed_u = (const float*)d_in[3];
  const float* embed_i = (const float*)d_in[4];
  const float* embed_r = (const float*)d_in[5];
  const float* gv_w1   = (const float*)d_in[6];
  const float* gv_b1   = (const float*)d_in[7];
  const float* gv_w2   = (const float*)d_in[8];
  const float* gv_b2   = (const float*)d_in[9];
  const float* gv_w3   = (const float*)d_in[10];
  const float* gv_b3   = (const float*)d_in[11];
  const float* att_w1  = (const float*)d_in[12];
  const float* att_b1  = (const float*)d_in[13];
  const float* att_w2  = (const float*)d_in[14];
  const float* att_b2  = (const float*)d_in[15];
  const float* att_w3  = (const float*)d_in[16];
  const float* att_b3  = (const float*)d_in[17];
  const float* wr1_w   = (const float*)d_in[18];
  const float* wr1_b   = (const float*)d_in[19];
  const float* wr2_w   = (const float*)d_in[20];
  const float* wr2_b   = (const float*)d_in[21];

  const int B = in_sizes[0];
  _Float16* wsT = (_Float16*)d_ws;

  prep_weights_k<<<dim3(32), dim3(256), 0, stream>>>(gv_w1, gv_w2, gv_w3,
                                                     att_w1, att_w2, wsT);
  item_fused_k<<<dim3(B), dim3(128), 0, stream>>>(
      nodes_v, neigh_u, neigh_r, embed_u, embed_i, embed_r,
      gv_b1, gv_b2, gv_b3,
      att_w1, att_b1, att_b2, att_w3, att_b3,
      wr1_w, wr1_b, wr2_w, wr2_b,
      wsT, (float*)d_out);
}
